// DotAttention_33174327394892
// MI455X (gfx1250) — matrix operand order repaired and checked by reading, not compiled, not run on hardware
//
#include <hip/hip_runtime.h>

typedef unsigned short u16;
typedef __attribute__((ext_vector_type(16))) __bf16 v16bf;
typedef __attribute__((ext_vector_type(8)))  float  v8f;

#define Bn 8
#define JXn 2048
#define JMn 2048
#define Dn 512
#define Hn 512

union FragB {
    uint4 q[2];
    v16bf v;
};

__device__ __forceinline__ u16 f2bf(float f) {
    unsigned int u = __float_as_uint(f);
    unsigned int lsb = (u >> 16) & 1u;
    u += 0x7fffu + lsb;
    return (u16)(u >> 16);
}

__device__ __forceinline__ v8f zero8() {
    v8f z;
#pragma unroll
    for (int i = 0; i < 8; ++i) z[i] = 0.0f;
    return z;
}

// ---------------------------------------------------------------------------
// CDNA5 async global->LDS copy (no VGPR staging, tracked by ASYNCcnt).
// 16 bytes per lane per issue; the IOFFSET immediate is added to BOTH the
// LDS and global addresses (ISA 15.18.3), so contiguous chunks need only one
// base address pair.
// ---------------------------------------------------------------------------
template <int OFF>
__device__ __forceinline__ void cp16_async(void* lds, const void* g) {
    asm volatile("global_load_async_to_lds_b128 %0, %1, off offset:%2"
                 :
                 : "v"((unsigned int)(unsigned long long)lds),
                   "v"((unsigned long long)g), "n"(OFF)
                 : "memory");
}
__device__ __forceinline__ void async_wait0() {
    asm volatile("s_wait_asynccnt 0" ::: "memory");
}

// 64B per lane, contiguous both sides
__device__ __forceinline__ void cp_g2l_4(u16* lds, const u16* g) {
    cp16_async<0>(lds, g);
    cp16_async<16>(lds, g);
    cp16_async<32>(lds, g);
    cp16_async<48>(lds, g);
}
// 128B per lane, contiguous both sides
__device__ __forceinline__ void cp_g2l_8(u16* lds, const u16* g) {
    cp16_async<0>(lds, g);
    cp16_async<16>(lds, g);
    cp16_async<32>(lds, g);
    cp16_async<48>(lds, g);
    cp16_async<64>(lds, g);
    cp16_async<80>(lds, g);
    cp16_async<96>(lds, g);
    cp16_async<112>(lds, g);
}

// ---------------------------------------------------------------------------
// Elementwise conversion kernels
// ---------------------------------------------------------------------------

// W f32 [K][N] -> WT bf16 [N][K]
__global__ void wtrans_kernel(const float* __restrict__ W, u16* __restrict__ WT,
                              int K, int N) {
    int idx = blockIdx.x * 256 + threadIdx.x;
    if (idx >= K * N) return;
    int k = idx / N, n = idx % N;
    WT[(size_t)n * K + k] = f2bf(W[idx]);
}

// X f32 -> bf16, same layout
__global__ void f2bf_kernel(const float* __restrict__ X, u16* __restrict__ Y,
                            size_t n) {
    size_t idx = (size_t)blockIdx.x * 256 + threadIdx.x;
    if (idx >= n) return;
    Y[idx] = f2bf(X[idx]);
}

// memory f32 [B][JM][D] -> memT bf16 [B][D][JM]
__global__ void memT_kernel(const float* __restrict__ M, u16* __restrict__ MT) {
    size_t idx = (size_t)blockIdx.x * 256 + threadIdx.x;
    if (idx >= (size_t)Bn * JMn * Dn) return;
    int b  = (int)(idx >> 20);              // JM*D = 2^20
    int r  = (int)(idx & ((1u << 20) - 1));
    int jm = r >> 9;                        // D = 512
    int d  = r & 511;
    MT[((size_t)b * Dn + d) * JMn + jm] = f2bf(M[idx]);
}

// res = concat(inputs, att): write bf16 copy + f32 copy (into d_out)
__global__ void pack_res_kernel(const float* __restrict__ inputs,
                                const float* __restrict__ att,
                                u16* __restrict__ resbf,
                                float* __restrict__ resf) {
    size_t idx = (size_t)blockIdx.x * 256 + threadIdx.x;
    if (idx >= (size_t)Bn * JXn * 2 * Dn) return;
    int m = (int)(idx >> 10);               // 2D = 1024
    int c = (int)(idx & 1023);
    float v = (c < Dn) ? inputs[(size_t)m * Dn + c]
                       : att[(size_t)m * Dn + (c - Dn)];
    resbf[idx] = f2bf(v);
    resf[idx]  = v;
}

// ---------------------------------------------------------------------------
// Generic bf16 GEMM:  C[M,N] = A[M,K] (row-major bf16) x B, with B given as
// BT[N,K] (row-major bf16, i.e. B column-major).  64x64 tile, 4 waves,
// double-buffered async global->LDS staging (copy overlaps WMMA).
// EPI==0: relu -> bf16 store.  EPI==1: out = resf * sigmoid(C) (f32, in place).
// ---------------------------------------------------------------------------
template <int EPI>
__global__ void __launch_bounds__(128)
gemm_bf16_kernel(const u16* __restrict__ A, const u16* __restrict__ BT,
                 u16* __restrict__ Cbf, float* __restrict__ Cf,
                 int M, int N, int K) {
    __shared__ u16 As[2 * 64 * 64];    // 16KB (двойной буфер)
    __shared__ u16 Bs[2 * 64 * 64];    // 16KB

    const int m0 = blockIdx.x * 64;
    const int n0 = blockIdx.y * 64;
    const int tid = threadIdx.x;
    const int wave = tid >> 5, lane = tid & 31;
    const int half = lane >> 4, ln = lane & 15;
    const int r = tid >> 1, c = (tid & 1) * 32;

    v8f acc[4];
#pragma unroll
    for (int t = 0; t < 4; ++t) acc[t] = zero8();

    const u16* gA = A + (size_t)(m0 + r) * K + c;
    const u16* gB = BT + (size_t)(n0 + r) * K + c;

    // prologue: stage k-piece 0 into buffer 0
    cp_g2l_4(As + r * 64 + c, gA);
    cp_g2l_4(Bs + r * 64 + c, gB);

    const int nk = K >> 6;
    for (int pc = 0; pc < nk; ++pc) {
        const u16* Ac = As + (pc & 1) * 4096;
        const u16* Bc = Bs + (pc & 1) * 4096;
        async_wait0();          // my copies into current buffer are done
        __syncthreads();        // everyone's copies done; next buffer is free
        if (pc + 1 < nk) {      // overlap next copy with current compute
            const int kn = (pc + 1) << 6;
            cp_g2l_4(As + ((pc + 1) & 1) * 4096 + r * 64 + c, gA + kn);
            cp_g2l_4(Bs + ((pc + 1) & 1) * 4096 + r * 64 + c, gB + kn);
        }
#pragma unroll
        for (int kk = 0; kk < 64; kk += 32) {
            FragB af, bfr[4];
            af.q[0] = *(const uint4*)(Ac + (16 * wave + ln) * 64 + kk + 16 * half);
            af.q[1] =
                *(const uint4*)(Ac + (16 * wave + ln) * 64 + kk + 8 + 16 * half);
#pragma unroll
            for (int t = 0; t < 4; ++t) {
                const uint4* bp =
                    (const uint4*)(Bc + (16 * t + ln) * 64 + kk + 16 * half);
                bfr[t].q[0] = bp[0];
                bfr[t].q[1] = bp[1];
            }
#pragma unroll
            for (int t = 0; t < 4; ++t)
                acc[t] = __builtin_amdgcn_wmma_f32_16x16x32_bf16(
                    false, af.v, false, bfr[t].v, (short)0, acc[t], false, false);
        }
    }

#pragma unroll
    for (int t = 0; t < 4; ++t) {
#pragma unroll
        for (int i = 0; i < 8; ++i) {
            const int row = m0 + 16 * wave + i + 8 * half;
            const int col = n0 + 16 * t + ln;
            float v = acc[t][i];
            if (EPI == 0) {
                v = fmaxf(v, 0.0f);
                Cbf[(size_t)row * N + col] = f2bf(v);
            } else {
                const float rr = Cf[(size_t)row * N + col];
                const float g = 1.0f / (1.0f + __expf(-v));
                Cf[(size_t)row * N + col] = rr * g;
            }
        }
    }
}

// ---------------------------------------------------------------------------
// Fused flash attention.
// grid = (JX/16, B), block = 128 (4 wave32).
// Each WG owns 16 JX rows; per JM chunk of 64:
//   wave w computes S slice cols [16w,16w+16)  (16 WMMA over H=512, Kt tiles
//     double-buffered through async-to-LDS so copies overlap WMMA)
//   cross-wave online softmax, P -> bf16 in LDS
//   wave w accumulates O cols [128w,128w+128)  (16 WMMA)
// ---------------------------------------------------------------------------
__global__ void __launch_bounds__(128)
attn_kernel(const u16* __restrict__ qb, const u16* __restrict__ kb,
            const u16* __restrict__ mT, const int* __restrict__ mask,
            float* __restrict__ att) {
    __shared__ u16 qs[16 * Hn];        // 16KB: q tile [16][512]
    __shared__ u16 stage[16384];       // 32KB: 2 x Kt[64][128]  or  MTs[512][32]
    __shared__ u16 Pbuf[16 * 64];      // 2KB : P tile [16][64]
    __shared__ float redmx[4][16];
    __shared__ float redsm[4][16];

    const int b = blockIdx.y;
    const int row0 = blockIdx.x * 16;
    const int tid = threadIdx.x;
    const int wave = tid >> 5, lane = tid & 31;
    const int half = lane >> 4, ln = lane & 15;
    const int kr = tid >> 1, kc = (tid & 1) * 64;   // Kt staging split

    // stage q tile: rows (b*JX+row0 .. +15), all 512 cols of bf16 q
    {
        const int base = tid * 64;           // ushort index into [16][512]
        const int r = base >> 9, c = base & 511;
        cp_g2l_8(qs + base, qb + ((size_t)(b * JXn + row0 + r) << 9) + c);
        async_wait0();
    }

    v8f Oacc[8];
#pragma unroll
    for (int t = 0; t < 8; ++t) Oacc[t] = zero8();
    float m_run[8], l_run[8];
#pragma unroll
    for (int i = 0; i < 8; ++i) { m_run[i] = -1e30f; l_run[i] = 0.0f; }

    const float scale = 0.044194173824159216f;  // 1/sqrt(512)

    for (int jm0 = 0; jm0 < JMn; jm0 += 64) {
        const u16* kRow = kb + ((size_t)(b * JMn + jm0 + kr) << 9) + kc;
        // ---- Phase A: S slice for this wave (pipelined Kt staging) --------
        v8f Sacc = zero8();
        __syncthreads();                       // stage free (Phase B done)
        cp_g2l_8(stage + kr * 128 + kc, kRow); // piece hp=0 -> buffer 0
        for (int hp = 0; hp < 4; ++hp) {
            const int h0 = hp * 128;
            const u16* Kt = stage + (hp & 1) * 8192;
            async_wait0();                     // my piece-hp copies done
            __syncthreads();                   // all copies done; other buf free
            if (hp < 3)                        // overlap next copy with WMMA
                cp_g2l_8(stage + ((hp + 1) & 1) * 8192 + kr * 128 + kc,
                         kRow + h0 + 128);
#pragma unroll
            for (int kg = 0; kg < 2; ++kg) {
                FragB af[2], bf2[2];
#pragma unroll
                for (int u = 0; u < 2; ++u) {
                    const int k0 = (kg * 2 + u) * 32;
                    af[u].q[0] =
                        *(const uint4*)(qs + (ln << 9) + h0 + k0 + 16 * half);
                    af[u].q[1] =
                        *(const uint4*)(qs + (ln << 9) + h0 + k0 + 8 + 16 * half);
                    const uint4* bp = (const uint4*)(Kt +
                        (16 * wave + ln) * 128 + k0 + 16 * half);
                    bf2[u].q[0] = bp[0];
                    bf2[u].q[1] = bp[1];
                }
#pragma unroll
                for (int u = 0; u < 2; ++u)
                    Sacc = __builtin_amdgcn_wmma_f32_16x16x32_bf16(
                        false, af[u].v, false, bf2[u].v, (short)0, Sacc, false,
                        false);
            }
        }

        // ---- mask + online softmax ----------------------------------------
        const int jmcol = jm0 + 16 * wave + ln;
        const int mk = mask[b * JMn + jmcol];
        float s[8], rmax[8];
#pragma unroll
        for (int i = 0; i < 8; ++i) {
            const float v = Sacc[i] * scale;
            s[i] = mk ? v : -1e30f;
            rmax[i] = s[i];
        }
#pragma unroll
        for (int d = 1; d < 16; d <<= 1)
#pragma unroll
            for (int i = 0; i < 8; ++i)
                rmax[i] = fmaxf(rmax[i], __shfl_xor(rmax[i], d, 32));
        if (ln == 0)
#pragma unroll
            for (int i = 0; i < 8; ++i) redmx[wave][half * 8 + i] = rmax[i];
        __syncthreads();

        float alpha[8], p[8], rsum[8];
#pragma unroll
        for (int i = 0; i < 8; ++i) {
            const float mc =
                fmaxf(fmaxf(redmx[0][half * 8 + i], redmx[1][half * 8 + i]),
                      fmaxf(redmx[2][half * 8 + i], redmx[3][half * 8 + i]));
            const float mnew = fmaxf(m_run[i], mc);
            alpha[i] = __expf(m_run[i] - mnew);
            m_run[i] = mnew;
            p[i] = mk ? __expf(s[i] - mnew) : 0.0f;
            rsum[i] = p[i];
        }
#pragma unroll
        for (int d = 1; d < 16; d <<= 1)
#pragma unroll
            for (int i = 0; i < 8; ++i) rsum[i] += __shfl_xor(rsum[i], d, 32);
        if (ln == 0)
#pragma unroll
            for (int i = 0; i < 8; ++i) redsm[wave][half * 8 + i] = rsum[i];
        // write P tile (bf16)
#pragma unroll
        for (int i = 0; i < 8; ++i)
            Pbuf[(i + 8 * half) * 64 + 16 * wave + ln] = f2bf(p[i]);
        __syncthreads();
#pragma unroll
        for (int i = 0; i < 8; ++i) {
            const float sc = redsm[0][half * 8 + i] + redsm[1][half * 8 + i] +
                             redsm[2][half * 8 + i] + redsm[3][half * 8 + i];
            l_run[i] = l_run[i] * alpha[i] + sc;
        }
        // rescale running O
#pragma unroll
        for (int t = 0; t < 8; ++t)
#pragma unroll
            for (int i = 0; i < 8; ++i) Oacc[t][i] *= alpha[i];

        // ---- Phase B: O += P @ memory -------------------------------------
        for (int mp = 0; mp < 2; ++mp) {
            __syncthreads();
            {   // stage MTs[512][32] from memT (async direct-to-LDS)
                const int d0 = tid * 4;
#pragma unroll
                for (int rr = 0; rr < 4; ++rr)
                    cp_g2l_4(stage + (d0 + rr) * 32,
                             mT + ((size_t)(b * Dn + d0 + rr) << 11) + jm0 +
                                 mp * 32);
                async_wait0();
            }
            __syncthreads();
            FragB af;
            af.q[0] = *(const uint4*)(Pbuf + ln * 64 + mp * 32 + 16 * half);
            af.q[1] = *(const uint4*)(Pbuf + ln * 64 + mp * 32 + 8 + 16 * half);
#pragma unroll
            for (int tg = 0; tg < 2; ++tg) {
                FragB bfr[4];
#pragma unroll
                for (int t = 0; t < 4; ++t) {
                    const uint4* bp = (const uint4*)(stage +
                        (128 * wave + 16 * (4 * tg + t) + ln) * 32 + 16 * half);
                    bfr[t].q[0] = bp[0];
                    bfr[t].q[1] = bp[1];
                }
#pragma unroll
                for (int t = 0; t < 4; ++t)
                    Oacc[4 * tg + t] = __builtin_amdgcn_wmma_f32_16x16x32_bf16(
                        false, af.v, false, bfr[t].v, (short)0, Oacc[4 * tg + t],
                        false, false);
            }
        }
    }

    // ---- epilogue: att = O / l ------------------------------------------
#pragma unroll
    for (int t = 0; t < 8; ++t)
#pragma unroll
        for (int i = 0; i < 8; ++i) {
            const int row = row0 + i + 8 * half;
            const int dcol = 128 * wave + 16 * t + ln;
            att[((size_t)(b * JXn + row) << 9) + dcol] = Oacc[t][i] / l_run[i];
        }
}

// ---------------------------------------------------------------------------
// Launch
// ---------------------------------------------------------------------------
extern "C" void kernel_launch(void* const* d_in, const int* in_sizes, int n_in,
                              void* d_out, int out_size, void* d_ws,
                              size_t ws_size, hipStream_t stream) {
    (void)in_sizes; (void)n_in; (void)out_size; (void)ws_size;
    const float* inputs = (const float*)d_in[0];
    const float* memory = (const float*)d_in[1];
    const int*   mask   = (const int*)d_in[2];
    const float* Wq     = (const float*)d_in[3];
    const float* Wk     = (const float*)d_in[4];
    const float* Wg     = (const float*)d_in[5];
    float* out = (float*)d_out;

    char* ws = (char*)d_ws;
    size_t off = 0;
    u16* WqT = (u16*)(ws + off); off += (size_t)Dn * Hn * 2;            // 512KB
    u16* WkT = (u16*)(ws + off); off += (size_t)Dn * Hn * 2;            // 512KB
    u16* WgT = (u16*)(ws + off); off += (size_t)2 * Dn * 2 * Dn * 2;    // 2MB
    u16* inb = (u16*)(ws + off); off += (size_t)Bn * JXn * Dn * 2;      // 16MB
    u16* meb = (u16*)(ws + off); off += (size_t)Bn * JMn * Dn * 2;      // 16MB
    u16* mTb = (u16*)(ws + off); off += (size_t)Bn * Dn * JMn * 2;      // 16MB
    u16* qb  = (u16*)(ws + off); off += (size_t)Bn * JXn * Hn * 2;      // 16MB
    u16* kb  = (u16*)(ws + off); off += (size_t)Bn * JMn * Hn * 2;      // 16MB
    float* attb = (float*)(ws + off); off += (size_t)Bn * JXn * Dn * 4; // 32MB
    u16* rsb = (u16*)(ws + off); off += (size_t)Bn * JXn * 2 * Dn * 2;  // 32MB

    const int Mrows = Bn * JXn;  // 16384

    // weight transposes -> bf16
    wtrans_kernel<<<(Dn * Hn + 255) / 256, 256, 0, stream>>>(Wq, WqT, Dn, Hn);
    wtrans_kernel<<<(Dn * Hn + 255) / 256, 256, 0, stream>>>(Wk, WkT, Dn, Hn);
    wtrans_kernel<<<(2 * Dn * 2 * Dn + 255) / 256, 256, 0, stream>>>(
        Wg, WgT, 2 * Dn, 2 * Dn);
    // activations -> bf16
    f2bf_kernel<<<(unsigned)(((size_t)Mrows * Dn + 255) / 256), 256, 0, stream>>>(
        inputs, inb, (size_t)Mrows * Dn);
    f2bf_kernel<<<(unsigned)(((size_t)Bn * JMn * Dn + 255) / 256), 256, 0, stream>>>(
        memory, meb, (size_t)Bn * JMn * Dn);
    memT_kernel<<<(unsigned)(((size_t)Bn * JMn * Dn + 255) / 256), 256, 0, stream>>>(
        memory, mTb);

    // q = relu(inputs @ Wq), k = relu(memory @ Wk)   [bf16 out]
    gemm_bf16_kernel<0><<<dim3(Mrows / 64, Hn / 64), 128, 0, stream>>>(
        inb, WqT, qb, nullptr, Mrows, Hn, Dn);
    gemm_bf16_kernel<0><<<dim3(Bn * JMn / 64, Hn / 64), 128, 0, stream>>>(
        meb, WkT, kb, nullptr, Bn * JMn, Hn, Dn);

    // fused masked-softmax attention: att = softmax(qk^T/sqrt(H)) @ memory
    attn_kernel<<<dim3(JXn / 16, Bn), 128, 0, stream>>>(qb, kb, mTb, mask, attb);

    // res = concat(inputs, att)  (bf16 for GEMM + f32 into d_out)
    pack_res_kernel<<<(unsigned)(((size_t)Mrows * 2 * Dn + 255) / 256), 256, 0,
                      stream>>>(inputs, attb, rsb, out);

    // out = res * sigmoid(res @ Wg)   (in place on d_out)
    gemm_bf16_kernel<1><<<dim3(Mrows / 64, 2 * Dn / 64), 128, 0, stream>>>(
        rsb, WgT, nullptr, out, Mrows, 2 * Dn, 2 * Dn);
}
